// InvNet_36833639530809
// MI455X (gfx1250) — compile-verified, matrix operand order checked by reading, not run on hardware
//
#include <hip/hip_runtime.h>
#include <hip/hip_bf16.h>
#include <math.h>

// ---------------------------------------------------------------------------
// InvNet loss on MI455X (gfx1250):
//   logits = inputs @ em^T / 0.05 ; log_softmax ; top-6 smoothing ; mean NLL
// bf16 WMMA GEMM fused with online softmax + register top-6 (packed keys).
// Tiling: block = 32 rows (A stripe 128KB LDS), 8 waves; each wave does a
// 2x2 grid of 16x16 tiles: per K-chunk 2 A ds-loads + 2 B global-loads feed
// 4 WMMAs on independent accumulators (1 b128-load-pair per WMMA).
// Top-6 candidates are kept as order-preserving uint keys with a 10-bit
// column-tile code in the low mantissa bits -> 96 VGPRs instead of 192.
// ---------------------------------------------------------------------------

typedef __attribute__((ext_vector_type(16))) __bf16 v16bf;
typedef __attribute__((ext_vector_type(8)))  float  v8f;
typedef __attribute__((ext_vector_type(4)))  unsigned int v4u;

#define N_ROWS 4096
#define DIMK   2048
#define NCLS   16384
#define INV_BETA 20.0f
#define MROWS  32              // rows per block
#define NBLOCKS (N_ROWS / MROWS)

// ---- f32 -> bf16 (round to nearest even) ----------------------------------
__global__ void f32_to_bf16_kernel(const float* __restrict__ src,
                                   unsigned short* __restrict__ dst, int n) {
  int i = blockIdx.x * blockDim.x + threadIdx.x;
  if (i < n) {
    unsigned int u = __float_as_uint(src[i]);
    unsigned int r = u + 0x7FFFu + ((u >> 16) & 1u);
    dst[i] = (unsigned short)(r >> 16);
  }
}

// CDNA5 16-bit A/B fragment: elems 0..7 = K[khalf*8+0..7], 8..15 = K[16+khalf*8+0..7]
__device__ __forceinline__ v16bf load_frag(const unsigned short* p) {
  union { v4u u[2]; v16bf v; } x;
  x.u[0] = *(const v4u*)(p);
  x.u[1] = *(const v4u*)(p + 16);
  return x.v;
}

__device__ __forceinline__ float bf16_to_f32(unsigned short h) {
  return __uint_as_float(((unsigned int)h) << 16);
}

// Monotone float->uint key; low 10 bits replaced by tile code.
__device__ __forceinline__ unsigned make_key(float x, unsigned scode) {
  unsigned u = __float_as_uint(x);
  unsigned k = ((int)u < 0) ? ~u : (u | 0x80000000u);
  return (k & ~0x3FFu) | scode;
}
__device__ __forceinline__ float key_value(unsigned k) {
  unsigned u = (k & 0x80000000u) ? (k & 0x7FFFFFFFu) : ~k;
  return __uint_as_float(u);
}

// ---- fused GEMM + online softmax + top-6 ----------------------------------
__global__ __launch_bounds__(256, 1)
void invnet_main_kernel(const unsigned short* __restrict__ in_b,
                        const unsigned short* __restrict__ em_b,
                        const int* __restrict__ targets,
                        const int* __restrict__ epoch_p,
                        float* __restrict__ partial) {
  extern __shared__ unsigned char smem_raw[];
  unsigned short* A   = (unsigned short*)smem_raw;                 // 131072 B
  float* topV   = (float*)(smem_raw + 131072);                     // 8*32*6 f
  int*   topI   = (int*)  (smem_raw + 131072 + 6144);              // 8*32*6 i
  float* redM   = (float*)(smem_raw + 131072 + 12288);             // 8*32 f
  float* redS   = (float*)(smem_raw + 131072 + 13312);             // 8*32 f
  float* rowLos = (float*)(smem_raw + 131072 + 14336);             // 32 f

  const int tid    = threadIdx.x;
  const int wave   = tid >> 5;
  const int lane   = tid & 31;
  const int lane16 = lane & 15;
  const int half   = lane >> 4;        // K-half group per WMMA layout
  const int row0   = blockIdx.x * MROWS;

  // Stage A-stripe (32 rows x 2048 K, bf16 = 128KB) into LDS once.
  {
    const v4u* src = (const v4u*)(in_b + (size_t)row0 * DIMK);
    v4u* dst = (v4u*)A;
    for (int i = tid; i < (MROWS * DIMK) / 8; i += 256) dst[i] = src[i];
  }
  __syncthreads();

  // Per-lane state: 16 row-slots (2 M-tiles x 8). slot = mt*8 + r,
  // global row = mt*16 + r + 8*half.
  float m[16], s[16];
  unsigned tk[96];
#pragma unroll
  for (int q = 0; q < 16; ++q) { m[q] = -INFINITY; s[q] = 0.0f; }
#pragma unroll
  for (int j = 0; j < 96; ++j) tk[j] = 0u;

  const unsigned short* a_ptr0 = A + lane16 * DIMK + half * 8;
  const unsigned short* a_ptr1 = a_ptr0 + 16 * DIMK;

  // 512 column-tile pairs (32 cols each); 8 waves stride -> 64 passes/wave.
  for (int pair = wave; pair < NCLS / 32; pair += 8) {
    const int n0 = pair << 5;
    const unsigned short* b_ptr0 =
        em_b + (size_t)(n0 + lane16) * DIMK + half * 8;
    const unsigned short* b_ptr1 = b_ptr0 + (size_t)16 * DIMK;

    v8f c00 = {0.f,0.f,0.f,0.f,0.f,0.f,0.f,0.f};
    v8f c01 = c00, c10 = c00, c11 = c00;
#pragma unroll 2
    for (int k0 = 0; k0 < DIMK; k0 += 32) {
      v16bf a0 = load_frag(a_ptr0 + k0);
      v16bf a1 = load_frag(a_ptr1 + k0);
      v16bf b0 = load_frag(b_ptr0 + k0);
      v16bf b1 = load_frag(b_ptr1 + k0);
      c00 = __builtin_amdgcn_wmma_f32_16x16x32_bf16(false, a0, false, b0, (short)0, c00, false, false);
      c10 = __builtin_amdgcn_wmma_f32_16x16x32_bf16(false, a1, false, b0, (short)0, c10, false, false);
      c01 = __builtin_amdgcn_wmma_f32_16x16x32_bf16(false, a0, false, b1, (short)0, c01, false, false);
      c11 = __builtin_amdgcn_wmma_f32_16x16x32_bf16(false, a1, false, b1, (short)0, c11, false, false);
    }

    // Epilogue: 4 tiles -> online softmax + keyed top-6 insertion.
    const unsigned scode_base = 1023u - (unsigned)(pair << 1);
#pragma unroll
    for (int nt = 0; nt < 2; ++nt) {
      const unsigned scode = scode_base - (unsigned)nt;
#pragma unroll
      for (int mt = 0; mt < 2; ++mt) {
        v8f c = (mt == 0) ? (nt == 0 ? c00 : c01) : (nt == 0 ? c10 : c11);
#pragma unroll
        for (int r = 0; r < 8; ++r) {
          const int q = mt * 8 + r;
          float x = c[r] * INV_BETA;
          float om = m[q];
          float nm = fmaxf(om, x);
          s[q] = s[q] * __expf(om - nm) + __expf(x - nm);
          m[q] = nm;
          unsigned ck = make_key(x, scode);
#pragma unroll
          for (int j = 0; j < 6; ++j) {
            bool gt = ck > tk[q * 6 + j];
            unsigned hi = gt ? ck : tk[q * 6 + j];
            unsigned lo = gt ? tk[q * 6 + j] : ck;
            tk[q * 6 + j] = hi;
            ck = lo;
          }
        }
      }
    }
  }

  // Softmax reduce within 16-lane groups, deposit per wave per row.
#pragma unroll
  for (int q = 0; q < 16; ++q) {
    float mm = m[q], ss = s[q];
    for (int off = 8; off >= 1; off >>= 1) {
      float om = __shfl_xor(mm, off, 16);
      float os = __shfl_xor(ss, off, 16);
      float nm = fmaxf(mm, om);
      ss = ss * __expf(mm - nm) + os * __expf(om - nm);
      mm = nm;
    }
    if (lane16 == 0) {
      int row = (q >> 3) * 16 + (q & 7) + 8 * half;
      redM[wave * MROWS + row] = mm;
      redS[wave * MROWS + row] = ss;
    }
  }

  // Wave-level top-6 per row: ballot max-extraction on packed keys.
#pragma unroll
  for (int q = 0; q < 16; ++q) {
    const int row = (q >> 3) * 16 + (q & 7) + 8 * half;
    for (int it = 0; it < 6; ++it) {
      unsigned v = tk[q * 6 + 0];
      unsigned gm = v;
      {
        unsigned o;
        o = (unsigned)__shfl_xor((int)gm, 8, 16); gm = (o > gm) ? o : gm;
        o = (unsigned)__shfl_xor((int)gm, 4, 16); gm = (o > gm) ? o : gm;
        o = (unsigned)__shfl_xor((int)gm, 2, 16); gm = (o > gm) ? o : gm;
        o = (unsigned)__shfl_xor((int)gm, 1, 16); gm = (o > gm) ? o : gm;
      }
      unsigned long long bal = __ballot(v == gm);
      unsigned bits = (unsigned)(bal >> (half * 16)) & 0xFFFFu;
      int winner16 = __ffs(bits) - 1;
      if (lane16 == 0) {
        int tcode = 1023 - (int)(gm & 0x3FFu);
        topV[(wave * MROWS + row) * 6 + it] = key_value(gm);
        topI[(wave * MROWS + row) * 6 + it] = (tcode << 4) | winner16;
      }
      if (lane16 == winner16) {  // pop my head
#pragma unroll
        for (int j = 0; j < 5; ++j) tk[q * 6 + j] = tk[q * 6 + j + 1];
        tk[q * 6 + 5] = 0u;
      }
    }
  }
  __syncthreads();

  // Final per-row merge + loss (threads 0..31, one row each).
  const int epoch = epoch_p[0];
  if (tid < MROWS) {
    const int row = tid;
    float mm = redM[row], ss = redS[row];
    for (int w = 1; w < 8; ++w) {
      float om = redM[w * MROWS + row], os = redS[w * MROWS + row];
      float nm = fmaxf(mm, om);
      ss = ss * __expf(mm - nm) + os * __expf(om - nm);
      mm = nm;
    }
    float logZ = mm + __logf(ss);

    float bv[6]; int bi[6];
#pragma unroll
    for (int j = 0; j < 6; ++j) { bv[j] = -INFINITY; bi[j] = -1; }
    for (int w = 0; w < 8; ++w) {
      for (int qq = 0; qq < 6; ++qq) {
        float cv = topV[(w * MROWS + row) * 6 + qq];
        int   ci = topI[(w * MROWS + row) * 6 + qq];
#pragma unroll
        for (int j = 0; j < 6; ++j) {
          bool gt = cv > bv[j];
          float kv = gt ? cv : bv[j]; int ki = gt ? ci : bi[j];
          float pv = gt ? bv[j] : cv; int pi = gt ? bi[j] : ci;
          bv[j] = kv; bi[j] = ki; cv = pv; ci = pi;
        }
      }
    }

    // target logit from the LDS A stripe and L2-resident bf16 em
    const int tgt = targets[row0 + row];
    const unsigned short* er = em_b + (size_t)tgt * DIMK;
    const unsigned short* ar = A + row * DIMK;
    float acc = 0.0f;
    for (int k = 0; k < DIMK; ++k)
      acc = fmaf(bf16_to_f32(ar[k]), bf16_to_f32(er[k]), acc);
    const float xt  = acc * INV_BETA;
    const float lpt = xt - logZ;

    float loss;
    if (epoch > 4) {  // KNN smoothing: weight 1/6 at top-6 (non-target), 1.0 at target
      float a6 = 0.0f;
#pragma unroll
      for (int j = 0; j < 6; ++j)
        if (bi[j] != tgt) a6 += (bv[j] - logZ);
      loss = -(a6 * (1.0f / 6.0f) + lpt);
    } else {
      loss = -lpt;
    }
    rowLos[row] = loss;
  }
  __syncthreads();
  if (tid == 0) {
    float sum = 0.0f;
    for (int j = 0; j < MROWS; ++j) sum += rowLos[j];
    partial[blockIdx.x] = sum;
  }
}

// ---- deterministic final reduction ----------------------------------------
__global__ void reduce_partials_kernel(const float* __restrict__ partial,
                                       float* __restrict__ out) {
  __shared__ float sh[NBLOCKS];
  sh[threadIdx.x] = partial[threadIdx.x];
  __syncthreads();
  for (int off = NBLOCKS / 2; off > 0; off >>= 1) {
    if (threadIdx.x < off) sh[threadIdx.x] += sh[threadIdx.x + off];
    __syncthreads();
  }
  if (threadIdx.x == 0) out[0] = sh[0] * (1.0f / (float)N_ROWS);
}

extern "C" void kernel_launch(void* const* d_in, const int* in_sizes, int n_in,
                              void* d_out, int out_size, void* d_ws, size_t ws_size,
                              hipStream_t stream) {
  const float* inputs  = (const float*)d_in[0];   // [4096, 2048] f32
  const int*   targets = (const int*)  d_in[1];   // [4096]
  const float* em      = (const float*)d_in[2];   // [16384, 2048] f32
  const int*   epoch   = (const int*)  d_in[3];   // scalar

  unsigned short* in_b = (unsigned short*)d_ws;
  unsigned short* em_b = (unsigned short*)((char*)d_ws + (size_t)16777216);
  float* partial       = (float*)((char*)d_ws + (size_t)16777216 + 67108864);

  const int nIn = N_ROWS * DIMK;   //  8,388,608
  const int nEm = NCLS  * DIMK;    // 33,554,432
  f32_to_bf16_kernel<<<(nIn + 255) / 256, 256, 0, stream>>>(inputs, in_b, nIn);
  f32_to_bf16_kernel<<<(nEm + 255) / 256, 256, 0, stream>>>(em, em_b, nEm);

  const size_t lds_bytes = 131072 + 6144 + 6144 + 1024 + 1024 + 128; // 145536
  hipFuncSetAttribute((const void*)invnet_main_kernel,
                      hipFuncAttributeMaxDynamicSharedMemorySize,
                      (int)lds_bytes);

  invnet_main_kernel<<<NBLOCKS, 256, lds_bytes, stream>>>(
      in_b, em_b, targets, epoch, partial);

  reduce_partials_kernel<<<1, NBLOCKS, 0, stream>>>(partial, (float*)d_out);
}
